// CausalSelfAttention_28192165331332
// MI455X (gfx1250) — matrix operand order repaired and checked by reading, not compiled, not run on hardware
//
#include <hip/hip_runtime.h>

// ---------------------------------------------------------------------------
// Problem constants (from reference): B=2, T=2048, D=1024, H=16, HKV=4, HD=64
// ---------------------------------------------------------------------------
constexpr int Bb   = 2;
constexpr int Ts   = 2048;
constexpr int Dd   = 1024;
constexpr int Hh   = 16;
constexpr int HKVv = 4;
constexpr int HDd  = 64;

typedef float        v8f   __attribute__((ext_vector_type(8)));
typedef __bf16       v16bf __attribute__((ext_vector_type(16)));
typedef __bf16       v8bf  __attribute__((ext_vector_type(8)));
typedef unsigned int u32x4 __attribute__((ext_vector_type(4)));
typedef int          i32x4 __attribute__((ext_vector_type(4)));
typedef int          i32x8 __attribute__((ext_vector_type(8)));

union BF16x16 { v16bf v; v8bf h[2]; };

#if defined(__has_builtin)
#if __has_builtin(__builtin_amdgcn_tensor_load_to_lds) && __has_builtin(__builtin_amdgcn_s_wait_tensorcnt)
#define CSA_HAVE_TDM 1
#endif
#endif
#ifndef CSA_HAVE_TDM
#define CSA_HAVE_TDM 0
#endif

// ---- wave32 cross-lane reductions over the 16-lane half that shares rows ----
__device__ __forceinline__ float rmax16(float v) {
#pragma unroll
  for (int m = 1; m < 16; m <<= 1) v = fmaxf(v, __shfl_xor(v, m, 32));
  return v;
}
__device__ __forceinline__ float rsum16(float v) {
#pragma unroll
  for (int m = 1; m < 16; m <<= 1) v += __shfl_xor(v, m, 32);
  return v;
}

#if CSA_HAVE_TDM
// ---------------------------------------------------------------------------
// TDM: DMA a 2D bf16 tile (32 elems wide x 64 rows, row stride = strideElems)
// from global memory into LDS at byte offset lds_off (rows packed 64B apart).
// D# layout per CDNA5 ISA ch.8 (group0 = 4 dwords, group1 = 8 dwords).
// ---------------------------------------------------------------------------
__device__ __forceinline__ void csa_tdm_tile_32x64(const __bf16* gptr,
                                                   unsigned lds_off,
                                                   unsigned strideElems) {
  unsigned long long ga = (unsigned long long)(uintptr_t)gptr;
  u32x4 g0;
  g0[0] = 1u;                                           // count=1, user mode
  g0[1] = lds_off;                                      // lds_addr (bytes)
  g0[2] = (unsigned)(ga & 0xFFFFFFFFu);                 // global_addr[31:0]
  g0[3] = (unsigned)((ga >> 32) & 0x01FFFFFFu)          // global_addr[56:32]
        | 0x80000000u;                                  // type=2 ("image")
  i32x8 g1;
  g1[0] = 0x00010000;            // workgroup_mask=0, data_size=1 (2 bytes)
  g1[1] = (int)(32u << 16);      // tensor_dim0[15:0]=32 at bits 63:48
  g1[2] = (int)(64u << 16);      // tensor_dim0[31:16]=0, tensor_dim1[15:0]=64
  g1[3] = (int)(32u << 16);      // tensor_dim1[31:16]=0, tile_dim0=32
  g1[4] = 64;                    // tile_dim1=64, tile_dim2=0 (2D)
  g1[5] = (int)strideElems;      // tensor_dim0_stride[31:0]
  g1[6] = 0;                     // stride0[47:32]=0, stride1 low=0
  g1[7] = 0;
  i32x4 z4 = {0, 0, 0, 0};
#if __clang_major__ >= 23
  i32x8 z8 = {0, 0, 0, 0, 0, 0, 0, 0};
  __builtin_amdgcn_tensor_load_to_lds(g0, g1, z4, z4, z8, 0);
#else
  __builtin_amdgcn_tensor_load_to_lds(g0, g1, z4, z4, 0);
#endif
}
#endif

// ---------------------------------------------------------------------------
// fp32 -> bf16 conversion
// ---------------------------------------------------------------------------
__global__ void csa_cvt_bf16(const float* __restrict__ in, __bf16* __restrict__ out, int n) {
  int i = blockIdx.x * blockDim.x + threadIdx.x;
  if (i < n) out[i] = (__bf16)in[i];
}

// ---------------------------------------------------------------------------
// C[M,N] = A[M,K] * W[N,K]^T   (bf16 in, fp32 out)
// 128-thread block (4 waves) -> 64x64 C macro-tile.
// Per K-step: 64x32 A tile + 64x32 B tile staged in LDS by the Tensor Data
// Mover (double-buffered, s_wait_tensorcnt), fragments read via ds_load_b128.
// A-frag (16-bit 16x32): lane half0 holds K 0..7 & 16..23, half1 K 8..15 & 24..31
// B-frag (16-bit 32x16): lane half0 holds K 0..7 & 16..23, half1 K 8..15 & 24..31
// ---------------------------------------------------------------------------
__global__ __launch_bounds__(128) void csa_wmma_gemm_bf16(
    const __bf16* __restrict__ A, const __bf16* __restrict__ W,
    float* __restrict__ C, int M, int N, int K) {
  __shared__ __attribute__((aligned(128))) __bf16 sA[2][64 * 32];
  __shared__ __attribute__((aligned(128))) __bf16 sB[2][64 * 32];

  const int lane = threadIdx.x & 31;
  const int wid  = threadIdx.x >> 5;          // 0..3 : wave's 16-row slice
  const int tilesN = N >> 6;
  const int bm = blockIdx.x / tilesN;
  const int bn = blockIdx.x % tilesN;
  if (bm * 64 >= M) return;

  const int lan  = lane & 15;
  const int half = lane >> 4;
  const int khA  = half * 8;
  const int khB  = half * 8;

  const __bf16* Abase = A + (size_t)(bm * 64) * K;
  const __bf16* Wbase = W + (size_t)(bn * 64) * K;

  v8f acc[4];
#pragma unroll
  for (int t = 0; t < 4; ++t)
#pragma unroll
    for (int i = 0; i < 8; ++i) acc[t][i] = 0.0f;

  auto stage_load = [&](int buf, int k0) {
#if CSA_HAVE_TDM
    if (wid == 0) {
      csa_tdm_tile_32x64(Abase + k0, (unsigned)(uintptr_t)&sA[buf][0], (unsigned)K);
      csa_tdm_tile_32x64(Wbase + k0, (unsigned)(uintptr_t)&sB[buf][0], (unsigned)K);
    }
#else
    // fallback: cooperative global->LDS copy (128 threads, 16 elems each side)
    const int tid = threadIdx.x;
    const int row = tid >> 1;
    const int col = (tid & 1) * 16;
    *(v8bf*)&sA[buf][row * 32 + col]     = *(const v8bf*)(Abase + (size_t)row * K + k0 + col);
    *(v8bf*)&sA[buf][row * 32 + col + 8] = *(const v8bf*)(Abase + (size_t)row * K + k0 + col + 8);
    *(v8bf*)&sB[buf][row * 32 + col]     = *(const v8bf*)(Wbase + (size_t)row * K + k0 + col);
    *(v8bf*)&sB[buf][row * 32 + col + 8] = *(const v8bf*)(Wbase + (size_t)row * K + k0 + col + 8);
#endif
  };

  const int nk = K >> 5;
  stage_load(0, 0);

  for (int ks = 0; ks < nk; ++ks) {
    const int cur = ks & 1;
    if (ks + 1 < nk) stage_load(1 - cur, (ks + 1) * 32);
#if CSA_HAVE_TDM
    if (wid == 0) {
      if (ks + 1 < nk) __builtin_amdgcn_s_wait_tensorcnt(2);  // oldest pair done
      else             __builtin_amdgcn_s_wait_tensorcnt(0);
    }
#endif
    __syncthreads();

    const __bf16* a_s = &sA[cur][0] + (size_t)(wid * 16 + lan) * 32;
    BF16x16 a;
    a.h[0] = *(const v8bf*)(a_s + khA);
    a.h[1] = *(const v8bf*)(a_s + khA + 16);
#pragma unroll
    for (int t = 0; t < 4; ++t) {
      const __bf16* b_s = &sB[cur][0] + (size_t)(16 * t + lan) * 32;
      BF16x16 b;
      b.h[0] = *(const v8bf*)(b_s + khB);
      b.h[1] = *(const v8bf*)(b_s + khB + 16);
      acc[t] = __builtin_amdgcn_wmma_f32_16x16x32_bf16(
          false, a.v, false, b.v, (short)0, acc[t], false, false);
    }
    __syncthreads();
  }

  // C/D layout: VGPR i -> row (i + 8*half), col = lane&15
#pragma unroll
  for (int t = 0; t < 4; ++t)
#pragma unroll
    for (int i = 0; i < 8; ++i) {
      int r = bm * 64 + wid * 16 + i + 8 * half;
      int c = bn * 64 + 16 * t + lan;
      C[(size_t)r * N + c] = acc[t][i];
    }
}

// ---------------------------------------------------------------------------
// RoPE on Q (+ per-head gain), fp32 in -> bf16 out, layout [B,T,H,HD]
// ---------------------------------------------------------------------------
__global__ void csa_rope_q(const float* __restrict__ qf, const float* __restrict__ gain,
                           __bf16* __restrict__ qb) {
  int idx = blockIdx.x * blockDim.x + threadIdx.x;
  if (idx >= Bb * Ts * Hh * (HDd / 2)) return;
  int j = idx & 31;
  int h = (idx >> 5) & (Hh - 1);
  int t = (idx >> 9) & (Ts - 1);
  int b = idx >> 20;
  size_t base = ((size_t)(b * Ts + t) * Hh + h) * HDd;
  float f = __expf(-(float)(2 * j) * (9.210340371976184f / (float)HDd));
  float ang = (float)t * f;
  float c = __cosf(ang), s = __sinf(ang);
  float x1 = qf[base + j], x2 = qf[base + j + HDd / 2];
  float g = gain[h];
  qb[base + j]           = (__bf16)((x1 * c - x2 * s) * g);
  qb[base + j + HDd / 2] = (__bf16)((x1 * s + x2 * c) * g);
}

// RoPE on K, fp32 in -> bf16 out, layout [B,T,HKV,HD]
__global__ void csa_rope_k(const float* __restrict__ kf, __bf16* __restrict__ kb) {
  int idx = blockIdx.x * blockDim.x + threadIdx.x;
  if (idx >= Bb * Ts * HKVv * (HDd / 2)) return;
  int j = idx & 31;
  int h = (idx >> 5) & (HKVv - 1);
  int t = (idx >> 7) & (Ts - 1);
  int b = idx >> 18;
  size_t base = ((size_t)(b * Ts + t) * HKVv + h) * HDd;
  float f = __expf(-(float)(2 * j) * (9.210340371976184f / (float)HDd));
  float ang = (float)t * f;
  float c = __cosf(ang), s = __sinf(ang);
  float x1 = kf[base + j], x2 = kf[base + j + HDd / 2];
  kb[base + j]           = (__bf16)(x1 * c - x2 * s);
  kb[base + j + HDd / 2] = (__bf16)(x1 * s + x2 * c);
}

// V: [B,T,HKV,HD] fp32 -> [B,HKV,HD,T] bf16 (so PV B-fragments load contiguous)
__global__ void csa_v_transpose(const float* __restrict__ vf, __bf16* __restrict__ vt) {
  int idx = blockIdx.x * blockDim.x + threadIdx.x;
  if (idx >= Bb * Ts * HKVv * HDd) return;
  int d   = idx & 63;
  int kvh = (idx >> 6) & (HKVv - 1);
  int t   = (idx >> 8) & (Ts - 1);
  int b   = idx >> 19;
  vt[((size_t)(b * HKVv + kvh) * HDd + d) * Ts + t] =
      (__bf16)vf[((size_t)(b * Ts + t) * HKVv + kvh) * HDd + d];
}

// ---------------------------------------------------------------------------
// Flash attention: one wave per (b, h, 16-row query tile).
// S = Q K^T (2 WMMAs / 16-key subtile), online softmax, O += P V (4 WMMAs per
// 32-key block). P goes through LDS to convert S C-layout -> A-layout.
// ---------------------------------------------------------------------------
__global__ __launch_bounds__(32) void csa_flash_attn(
    const __bf16* __restrict__ qb,   // [B,T,H,HD]
    const __bf16* __restrict__ kb,   // [B,T,HKV,HD]
    const __bf16* __restrict__ vt,   // [B,HKV,HD,T]
    __bf16* __restrict__ ob) {       // [B,T,H*HD]
  __shared__ __attribute__((aligned(16))) __bf16 sP[16 * 32];

  const int lane = threadIdx.x & 31;
  const int qt   = blockIdx.x & (Ts / 16 - 1);
  const int h    = (blockIdx.x >> 7) & (Hh - 1);
  const int b    = blockIdx.x >> 11;
  const int kvh  = h >> 2;                 // G = H/HKV = 4
  const int q0   = qt * 16;
  const int lan  = lane & 15;
  const int half = lane >> 4;
  const int khA  = half * 8;
  const int khB  = half * 8;

  BF16x16 qa[2];
  {
    const __bf16* qrow = qb + ((size_t)(b * Ts + q0 + lan) * Hh + h) * HDd;
#pragma unroll
    for (int c = 0; c < 2; ++c) {
      qa[c].h[0] = *(const v8bf*)(qrow + 32 * c + khA);
      qa[c].h[1] = *(const v8bf*)(qrow + 32 * c + khA + 16);
    }
  }

  v8f oacc[4];
  v8f mrow, lrow;
#pragma unroll
  for (int i = 0; i < 8; ++i) { mrow[i] = -1e30f; lrow[i] = 0.0f; }
#pragma unroll
  for (int t = 0; t < 4; ++t)
#pragma unroll
    for (int i = 0; i < 8; ++i) oacc[t][i] = 0.0f;

  const int qend = q0 + 15;
  const float scale = 0.125f;  // 1/sqrt(64)

  for (int s0 = 0; s0 <= qend; s0 += 32) {
    v8f sc[2];
#pragma unroll
    for (int sub = 0; sub < 2; ++sub) {
      const int n0 = s0 + 16 * sub;
      const __bf16* krow = kb + ((size_t)(b * Ts + n0 + lan) * HKVv + kvh) * HDd;
      BF16x16 k0f, k1f;
      k0f.h[0] = *(const v8bf*)(krow + khB);
      k0f.h[1] = *(const v8bf*)(krow + khB + 16);
      k1f.h[0] = *(const v8bf*)(krow + 32 + khB);
      k1f.h[1] = *(const v8bf*)(krow + 32 + khB + 16);
      v8f s;
#pragma unroll
      for (int i = 0; i < 8; ++i) s[i] = 0.0f;
      s = __builtin_amdgcn_wmma_f32_16x16x32_bf16(false, qa[0].v, false, k0f.v,
                                                  (short)0, s, false, false);
      s = __builtin_amdgcn_wmma_f32_16x16x32_bf16(false, qa[1].v, false, k1f.v,
                                                  (short)0, s, false, false);
#pragma unroll
      for (int i = 0; i < 8; ++i) {
        int qrow_i = q0 + i + 8 * half;
        int spos   = n0 + lan;
        sc[sub][i] = (spos <= qrow_i) ? s[i] * scale : -1e30f;
      }
    }

    v8f mnew, alpha, p0, p1;
#pragma unroll
    for (int i = 0; i < 8; ++i) {
      float bm = rmax16(fmaxf(sc[0][i], sc[1][i]));
      float mn = fmaxf(mrow[i], bm);
      mnew[i]  = mn;
      alpha[i] = __expf(mrow[i] - mn);
      mrow[i]  = mn;
    }
#pragma unroll
    for (int i = 0; i < 8; ++i) {
      p0[i] = __expf(sc[0][i] - mnew[i]);
      p1[i] = __expf(sc[1][i] - mnew[i]);
      lrow[i] = lrow[i] * alpha[i] + rsum16(p0[i] + p1[i]);
    }
#pragma unroll
    for (int t = 0; t < 4; ++t)
#pragma unroll
      for (int i = 0; i < 8; ++i) oacc[t][i] *= alpha[i];

    __syncthreads();
#pragma unroll
    for (int i = 0; i < 8; ++i) {
      int r = i + 8 * half;
      sP[r * 32 + lan]      = (__bf16)p0[i];
      sP[r * 32 + 16 + lan] = (__bf16)p1[i];
    }
    __syncthreads();
    BF16x16 pa;
    pa.h[0] = *(const v8bf*)(sP + lan * 32 + khA);
    pa.h[1] = *(const v8bf*)(sP + lan * 32 + khA + 16);

    const __bf16* vbase = vt + ((size_t)(b * HKVv + kvh) * HDd) * Ts + s0;
#pragma unroll
    for (int t = 0; t < 4; ++t) {
      const __bf16* vcol = vbase + (size_t)(16 * t + lan) * Ts;
      BF16x16 vb;
      vb.h[0] = *(const v8bf*)(vcol + khB);
      vb.h[1] = *(const v8bf*)(vcol + khB + 16);
      oacc[t] = __builtin_amdgcn_wmma_f32_16x16x32_bf16(
          false, pa.v, false, vb.v, (short)0, oacc[t], false, false);
    }
  }

#pragma unroll
  for (int t = 0; t < 4; ++t)
#pragma unroll
    for (int i = 0; i < 8; ++i) {
      int r = q0 + i + 8 * half;
      int d = 16 * t + lan;
      float val = oacc[t][i] / lrow[i];
      ob[((size_t)(b * Ts + r) * Hh + h) * HDd + d] = (__bf16)val;
    }
}

// ---------------------------------------------------------------------------
// Host-side orchestration
// ---------------------------------------------------------------------------
extern "C" void kernel_launch(void* const* d_in, const int* in_sizes, int n_in,
                              void* d_out, int out_size, void* d_ws, size_t ws_size,
                              hipStream_t stream) {
  const float* x  = (const float*)d_in[0];
  const float* Wq = (const float*)d_in[1];
  const float* Wk = (const float*)d_in[2];
  const float* Wv = (const float*)d_in[3];
  const float* Wp = (const float*)d_in[4];
  const float* qg = (const float*)d_in[5];

  char* ws = (char*)d_ws;
  size_t off = 0;
  auto alloc = [&](size_t bytes) -> void* {
    off = (off + 255) & ~(size_t)255;
    void* p = ws + off;
    off += bytes;
    return p;
  };

  const int BT = Bb * Ts;                      // 4096
  __bf16* xb  = (__bf16*)alloc((size_t)BT * Dd * 2);
  __bf16* wqb = (__bf16*)alloc((size_t)Dd * Dd * 2);
  __bf16* wkb = (__bf16*)alloc((size_t)(HKVv * HDd) * Dd * 2);
  __bf16* wvb = (__bf16*)alloc((size_t)(HKVv * HDd) * Dd * 2);
  __bf16* wpb = (__bf16*)alloc((size_t)Dd * Dd * 2);
  float*  qf  = (float*)alloc((size_t)BT * Dd * 4);
  float*  kf  = (float*)alloc((size_t)BT * (HKVv * HDd) * 4);
  float*  vf  = (float*)alloc((size_t)BT * (HKVv * HDd) * 4);
  __bf16* qb  = (__bf16*)alloc((size_t)BT * Dd * 2);
  __bf16* kb  = (__bf16*)alloc((size_t)BT * (HKVv * HDd) * 2);
  __bf16* vtb = (__bf16*)alloc((size_t)Bb * HKVv * HDd * Ts * 2);
  __bf16* ob  = (__bf16*)alloc((size_t)BT * Dd * 2);
  (void)ws_size; (void)n_in; (void)in_sizes; (void)out_size;

  const int nx  = BT * Dd;
  const int nwq = Dd * Dd;
  const int nwk = HKVv * HDd * Dd;

  // 1) bf16 conversions
  csa_cvt_bf16<<<(nx + 255) / 256, 256, 0, stream>>>(x, xb, nx);
  csa_cvt_bf16<<<(nwq + 255) / 256, 256, 0, stream>>>(Wq, wqb, nwq);
  csa_cvt_bf16<<<(nwk + 255) / 256, 256, 0, stream>>>(Wk, wkb, nwk);
  csa_cvt_bf16<<<(nwk + 255) / 256, 256, 0, stream>>>(Wv, wvb, nwk);
  csa_cvt_bf16<<<(nwq + 255) / 256, 256, 0, stream>>>(Wp, wpb, nwq);

  // 2) projections: q = x Wq^T, k = x Wk^T, v = x Wv^T   (64x64 tile / block)
  csa_wmma_gemm_bf16<<<(BT / 64) * (Dd / 64), 128, 0, stream>>>(xb, wqb, qf, BT, Dd, Dd);
  csa_wmma_gemm_bf16<<<(BT / 64) * ((HKVv * HDd) / 64), 128, 0, stream>>>(xb, wkb, kf, BT, HKVv * HDd, Dd);
  csa_wmma_gemm_bf16<<<(BT / 64) * ((HKVv * HDd) / 64), 128, 0, stream>>>(xb, wvb, vf, BT, HKVv * HDd, Dd);

  // 3) RoPE + gain, V transpose
  csa_rope_q<<<(Bb * Ts * Hh * (HDd / 2)) / 256, 256, 0, stream>>>(qf, qg, qb);
  csa_rope_k<<<(Bb * Ts * HKVv * (HDd / 2)) / 256, 256, 0, stream>>>(kf, kb);
  csa_v_transpose<<<(Bb * Ts * HKVv * HDd) / 256, 256, 0, stream>>>(vf, vtb);

  // 4) flash attention: one wave per (b, h, 16-row q tile)
  csa_flash_attn<<<Bb * Hh * (Ts / 16), 32, 0, stream>>>(qb, kb, vtb, ob);

  // 5) output projection into d_out (fp32)
  csa_wmma_gemm_bf16<<<(BT / 64) * (Dd / 64), 128, 0, stream>>>(ob, wpb, (float*)d_out, BT, Dd, Dd);
}
